// MLAttention_82660940579318
// MI455X (gfx1250) — compile-verified, matrix operand order checked by reading, not run on hardware
//
#include <hip/hip_runtime.h>
#include <hip/hip_bf16.h>

// ---------------------------------------------------------------------------
// MLA + top-k indexer for MI455X (gfx1250), wave32, bf16 WMMA w/ fp32 accum.
// TDM (tensor_load_to_lds) staging for bf16 operand tiles where available.
// ---------------------------------------------------------------------------

typedef __bf16 bf16;
typedef __attribute__((ext_vector_type(16))) __bf16 bf16x16;
typedef __attribute__((ext_vector_type(8)))  float  f32x8;
typedef __attribute__((ext_vector_type(4)))  unsigned u32x4;
typedef __attribute__((ext_vector_type(8)))  int      i32x8;
typedef __attribute__((ext_vector_type(4)))  int      i32x4;

union AFrag { bf16x16 v; uint4 q[2]; bf16 e[16]; };
union CFrag { f32x8  v; float e[8]; };

constexpr int B_ = 2, S_ = 2048, HID_ = 2048;
constexpr int H_ = 16, QK_NOPE_ = 128, ROPE_ = 64, QK_HEAD_ = 192, V_HEAD_ = 128;
constexpr int Q_RANK_ = 1024, KV_RANK_ = 512;
constexpr int IDX_H_ = 16, IDX_D_ = 128, TOPK_ = 512;
constexpr float kNEG = -1000000000.0f;

#if defined(__has_builtin)
#if __has_builtin(__builtin_amdgcn_tensor_load_to_lds) && \
    __has_builtin(__builtin_amdgcn_s_wait_tensorcnt)
#define HAVE_TDM 1
#endif
#endif
#ifndef HAVE_TDM
#define HAVE_TDM 0
#endif

__device__ __forceinline__ f32x8 wmma_bf16(bf16x16 a, bf16x16 b, f32x8 c) {
  return __builtin_amdgcn_wmma_f32_16x16x32_bf16(false, a, false, b, (short)0, c,
                                                 false, false);
}

#if HAVE_TDM
// LDS aperture keeps the byte offset in addr[31:0] (ISA 10.2): truncate flat ptr.
__device__ __forceinline__ unsigned lds_byte_addr(const void* p) {
  return (unsigned)(unsigned long long)p;
}

// Issue a 2D TDM tile load of bf16 data: tile1 rows x tile0 elements,
// source rows separated by stride0 elements. D# per CDNA5 ISA 8.3/8.4.
// This toolchain exposes the 6-arg builtin:
//   (u32x4 g0, i32x8 g1, i32x4 g2, i32x4 g3, i32x8 ext, i32 cpol)
__device__ __forceinline__ void tdm_load_2d_bf16(const void* gsrc, unsigned lds_addr,
                                                 unsigned tensor_dim0, unsigned tensor_dim1,
                                                 unsigned tile0, unsigned tile1,
                                                 unsigned long long stride0)
{
  const unsigned long long ga = (unsigned long long)gsrc;
  u32x4 g0;
  g0.x = 1u;                                              // count=1 (valid), user mode
  g0.y = lds_addr;                                        // LDS byte address
  g0.z = (unsigned)(ga & 0xffffffffull);                  // global_addr[31:0]
  g0.w = (unsigned)((ga >> 32) & 0x1ffffffull)            // global_addr[56:32]
         | (2u << 30);                                    // type = 2 ("image")
  i32x8 g1;
  g1[0] = (int)(1u << 16);                                // data_size=1 (2 bytes)
  g1[1] = (int)((tensor_dim0 & 0xffffu) << 16);           // tensor_dim0[15:0]
  g1[2] = (int)((tensor_dim0 >> 16) |
                ((tensor_dim1 & 0xffffu) << 16));         // dim0[31:16] | dim1[15:0]
  g1[3] = (int)(((tensor_dim1 >> 16) & 0xffffu) |
                (tile0 << 16));                           // dim1[31:16] | tile_dim0
  g1[4] = (int)(tile1 & 0xffffu);                         // tile_dim1 (tile_dim2=0)
  g1[5] = (int)(stride0 & 0xffffffffull);                 // dim0_stride[31:0]
  g1[6] = (int)((stride0 >> 32) & 0xffffull);             // dim0_stride[47:32]
  g1[7] = 0;                                              // dim1_stride unused (2D)
  const i32x4 z4 = {0, 0, 0, 0};
  const i32x8 z8 = {0, 0, 0, 0, 0, 0, 0, 0};
  __builtin_amdgcn_tensor_load_to_lds(g0, g1, z4, z4, z8, 0);
}
#endif

// ---------------------------------------------------------------------------
// Generic GEMM: C[M,N] = A[M,K] * B[K,N]; fp32 in/out, bf16 WMMA inside.
// M % 128 == 0, K % 32 == 0 assumed; N guarded.
// Block = 256 threads (8 waves as 2x4), tile 128x128x32.
// ---------------------------------------------------------------------------
__global__ __launch_bounds__(256)
void gemm_f32_bf16wmma(const float* __restrict__ A, const float* __restrict__ Bm,
                       float* __restrict__ C, int M, int N, int K)
{
  __shared__ bf16 As[128 * 32];   // [m][k]
  __shared__ bf16 Bst[128 * 32];  // transposed: [n][k]

  const int tid  = threadIdx.x;
  const int lane = tid & 31;
  const int wave = tid >> 5;
  const int l15  = lane & 15;
  const int kg   = lane >> 4;
  const int wm   = (wave >> 2) * 64;  // 0 / 64
  const int wn   = (wave & 3) * 32;   // 0 / 32 / 64 / 96
  const int m0   = blockIdx.y * 128;
  const int n0   = blockIdx.x * 128;

  CFrag acc[4][2] = {};

  for (int k0 = 0; k0 < K; k0 += 32) {
    // ---- stage A tile (128x32) as bf16 ----
    {
      const int row = tid >> 1;
      const int kc  = (tid & 1) * 16;
      const float* src = A + (size_t)(m0 + row) * K + k0 + kc;
      if (k0 + 32 < K) __builtin_prefetch(src + 32, 0, 1);
      bf16* dst = As + row * 32 + kc;
#pragma unroll
      for (int i = 0; i < 16; i += 4) {
        float4 f = *reinterpret_cast<const float4*>(src + i);
        dst[i + 0] = (bf16)f.x; dst[i + 1] = (bf16)f.y;
        dst[i + 2] = (bf16)f.z; dst[i + 3] = (bf16)f.w;
      }
    }
    // ---- stage B tile transposed (Bst[n][k]) ----
    {
      const int kr = tid >> 3;          // 0..31
      const int nc = (tid & 7) * 16;    // 0..112
      const float* src = Bm + (size_t)(k0 + kr) * N + n0 + nc;
#pragma unroll
      for (int i = 0; i < 16; ++i) {
        const int n = n0 + nc + i;
        const float f = (n < N) ? src[i] : 0.0f;
        Bst[(nc + i) * 32 + kr] = (bf16)f;
      }
    }
    __syncthreads();

    AFrag afr[4], bfr[2];
#pragma unroll
    for (int ms = 0; ms < 4; ++ms) {
      const bf16* p = As + (wm + ms * 16 + l15) * 32;
      afr[ms].q[0] = *reinterpret_cast<const uint4*>(p + 8 * kg);
      afr[ms].q[1] = *reinterpret_cast<const uint4*>(p + 16 + 8 * kg);
    }
#pragma unroll
    for (int ns = 0; ns < 2; ++ns) {
      const bf16* p = Bst + (wn + ns * 16 + l15) * 32;
      bfr[ns].q[0] = *reinterpret_cast<const uint4*>(p + 8 * kg);
      bfr[ns].q[1] = *reinterpret_cast<const uint4*>(p + 16 + 8 * kg);
    }
#pragma unroll
    for (int ms = 0; ms < 4; ++ms)
#pragma unroll
      for (int ns = 0; ns < 2; ++ns)
        acc[ms][ns].v = wmma_bf16(afr[ms].v, bfr[ns].v, acc[ms][ns].v);
    __syncthreads();
  }

#pragma unroll
  for (int ms = 0; ms < 4; ++ms)
#pragma unroll
    for (int ns = 0; ns < 2; ++ns)
#pragma unroll
      for (int r = 0; r < 8; ++r) {
        const int m = m0 + wm + ms * 16 + r + 8 * kg;
        const int n = n0 + wn + ns * 16 + l15;
        if (n < N) C[(size_t)m * N + n] = acc[ms][ns].e[r];
      }
}

// ---------------------------------------------------------------------------
// RMSNorm: one block (256 thr) per row. dim/256 elements per thread.
// ---------------------------------------------------------------------------
__global__ __launch_bounds__(256)
void rmsnorm_kernel(const float* __restrict__ in, const float* __restrict__ w,
                    float* __restrict__ out, int dim, int in_stride, int out_stride)
{
  const int row = blockIdx.x;
  const int tid = threadIdx.x;
  const float* x = in + (size_t)row * in_stride;
  float* y = out + (size_t)row * out_stride;
  __shared__ float red[256];

  float loc[4];
  const int n = dim / 256;
  float ss = 0.0f;
  for (int i = 0; i < n; ++i) { loc[i] = x[tid + i * 256]; ss += loc[i] * loc[i]; }
  red[tid] = ss; __syncthreads();
  for (int st = 128; st > 0; st >>= 1) {
    if (tid < st) red[tid] += red[tid + st];
    __syncthreads();
  }
  const float inv = rsqrtf(red[0] / (float)dim + 1e-6f);
  for (int i = 0; i < n; ++i) {
    const int c = tid + i * 256;
    y[c] = w[c] * loc[i] * inv;
  }
}

// ---------------------------------------------------------------------------
// RoPE on k_pe: ckv[row][512..575] -> kpe[row][0..63]
// ---------------------------------------------------------------------------
__global__ void rope_kpe_kernel(const float* __restrict__ ckv,
                                const float* __restrict__ cosb,
                                const float* __restrict__ sinb,
                                float* __restrict__ kpe)
{
  const int row = blockIdx.x;
  const int d = threadIdx.x;  // 0..63
  const float* x = ckv + (size_t)row * 576 + 512;
  const float xv = x[d];
  const float rot = (d < 32) ? -x[d + 32] : x[d - 32];
  kpe[(size_t)row * 64 + d] =
      xv * cosb[(size_t)row * 64 + d] + rot * sinb[(size_t)row * 64 + d];
}

// ---------------------------------------------------------------------------
// Build query bf16 [B,H,S,192] from q_full [B,S,H*192] (RoPE on last 64)
// ---------------------------------------------------------------------------
__global__ void build_query_kernel(const float* __restrict__ q_full,
                                   const float* __restrict__ cosb,
                                   const float* __restrict__ sinb,
                                   bf16* __restrict__ query)
{
  const int bsh = blockIdx.x;
  const int h = bsh % H_;
  const int row = bsh / H_;            // b*S + s
  const int b = row / S_, s = row % S_;
  const int d = threadIdx.x;           // 0..191
  const float* src = q_full + (size_t)row * (H_ * QK_HEAD_) + h * QK_HEAD_;
  float v;
  if (d < 128) {
    v = src[d];
  } else {
    const int dd = d - 128;
    const float x = src[128 + dd];
    const float rot = (dd < 32) ? -src[128 + dd + 32] : src[128 + dd - 32];
    v = x * cosb[(size_t)row * 64 + dd] + rot * sinb[(size_t)row * 64 + dd];
  }
  query[(((size_t)b * H_ + h) * S_ + s) * 192 + d] = (bf16)v;
}

// ---------------------------------------------------------------------------
// Build keyt bf16 [B,H,S,192] and value bf16 [B,H,S,128]
// from kv_full [B,S,H*256] and kpe [B,S,64]
// ---------------------------------------------------------------------------
__global__ void build_keyval_kernel(const float* __restrict__ kv_full,
                                    const float* __restrict__ kpe,
                                    bf16* __restrict__ keyt,
                                    bf16* __restrict__ value)
{
  const int bsh = blockIdx.x;
  const int h = bsh % H_;
  const int row = bsh / H_;
  const int b = row / S_, s = row % S_;
  const int d = threadIdx.x;  // 0..255
  const float* src = kv_full + (size_t)row * (H_ * 256) + h * 256;
  const size_t kbase = ((size_t)b * H_ + h) * S_ + s;
  if (d < 192) {
    const float v = (d < 128) ? src[d] : kpe[(size_t)row * 64 + (d - 128)];
    keyt[kbase * 192 + d] = (bf16)v;
  }
  if (d < 128) value[kbase * 128 + d] = (bf16)src[128 + d];
}

// ---------------------------------------------------------------------------
// Build qi bf16 [B,S,16,128] from qi_full (RoPE on first 64 of each head)
// ---------------------------------------------------------------------------
__global__ void build_qi_kernel(const float* __restrict__ qi_full,
                                const float* __restrict__ cosb,
                                const float* __restrict__ sinb,
                                bf16* __restrict__ qi)
{
  const int bsh = blockIdx.x;
  const int h = bsh % 16;
  const int row = bsh / 16;
  const int d = threadIdx.x;  // 0..127
  const float* src = qi_full + (size_t)row * 2048 + h * 128;
  float v;
  if (d < 64) {
    const float x = src[d];
    const float rot = (d < 32) ? -src[d + 32] : src[d - 32];
    v = x * cosb[(size_t)row * 64 + d] + rot * sinb[(size_t)row * 64 + d];
  } else {
    v = src[d];
  }
  qi[(size_t)row * 2048 + h * 128 + d] = (bf16)v;
}

// ---------------------------------------------------------------------------
// Build ki bf16 [B,S,128]: layernorm + RoPE(first 64)
// ---------------------------------------------------------------------------
__global__ __launch_bounds__(128)
void build_ki_kernel(const float* __restrict__ ki_tmp,
                     const float* __restrict__ lnw, const float* __restrict__ lnb,
                     const float* __restrict__ cosb, const float* __restrict__ sinb,
                     bf16* __restrict__ ki)
{
  const int row = blockIdx.x;
  const int d = threadIdx.x;  // 0..127
  __shared__ float red[128];
  __shared__ float xs[128];

  const float x = ki_tmp[(size_t)row * 128 + d];
  red[d] = x; __syncthreads();
  for (int st = 64; st > 0; st >>= 1) {
    if (d < st) red[d] += red[d + st];
    __syncthreads();
  }
  const float mean = red[0] / 128.0f;
  __syncthreads();
  const float c = x - mean;
  red[d] = c * c; __syncthreads();
  for (int st = 64; st > 0; st >>= 1) {
    if (d < st) red[d] += red[d + st];
    __syncthreads();
  }
  const float var = red[0] / 128.0f;
  const float y = c * rsqrtf(var + 1e-6f) * lnw[d] + lnb[d];
  xs[d] = y; __syncthreads();
  float v;
  if (d < 64) {
    const float rot = (d < 32) ? -xs[d + 32] : xs[d - 32];
    v = y * cosb[(size_t)row * 64 + d] + rot * sinb[(size_t)row * 64 + d];
  } else {
    v = y;
  }
  ki[(size_t)row * 128 + d] = (bf16)v;
}

// ---------------------------------------------------------------------------
// Indexer scores: out[b][s][t] = sum_h relu(qi[b,s,h,:].ki[b,t,:]*IDX_D^-.5)
//                               * (wts[b,s,h]*IDX_H^-.5)   + causal
// Tile: 16 s-rows x 128 t-cols per block; wave w owns t-subtile w*16.
// ki tile (once) and qi head tiles (per h) staged via TDM when available.
// ---------------------------------------------------------------------------
__global__ __launch_bounds__(256)
void idx_scores_kernel(const bf16* __restrict__ qi, const bf16* __restrict__ ki,
                       const float* __restrict__ wts, float* __restrict__ out)
{
  const int b = blockIdx.z;
  const int s0 = blockIdx.y * 16;
  const int t0 = blockIdx.x * 128;
  const int tid = threadIdx.x;
  const int lane = tid & 31;
  const int w = tid >> 5;
  const int l15 = lane & 15;
  const int kg = lane >> 4;

  __shared__ bf16 kis[128 * 128];  // [t][d]
  __shared__ bf16 qis[16 * 128];   // [s][d] (per head)
  __shared__ float wl[16 * 16];    // [s][h]

#if HAVE_TDM
  if (tid == 0) {
    tdm_load_2d_bf16(ki + ((size_t)b * S_ + t0) * 128, lds_byte_addr(kis),
                     /*dim0=*/128u, /*dim1=*/(unsigned)(S_ - t0),
                     /*tile0=*/128u, /*tile1=*/128u, /*stride0=*/128ull);
    __builtin_amdgcn_s_wait_tensorcnt(0);
  }
#else
  {  // stage ki tile
    const int tr = tid >> 1;
    const int dc = (tid & 1) * 64;
    const bf16* src = ki + ((size_t)b * S_ + t0 + tr) * 128 + dc;
    bf16* dst = kis + tr * 128 + dc;
#pragma unroll
    for (int i = 0; i < 64; i += 8)
      *reinterpret_cast<uint4*>(dst + i) = *reinterpret_cast<const uint4*>(src + i);
  }
#endif
  {  // stage wts (pre-scaled by IDX_H^-0.5 = 0.25)
    const int sr = tid >> 4;
    const int h = tid & 15;
    wl[sr * 16 + h] = wts[((size_t)b * S_ + s0 + sr) * 16 + h] * 0.25f;
  }
  __syncthreads();

  AFrag bfr[4];
#pragma unroll
  for (int kc = 0; kc < 4; ++kc) {
    const bf16* p = kis + (w * 16 + l15) * 128 + kc * 32;
    bfr[kc].q[0] = *reinterpret_cast<const uint4*>(p + 8 * kg);
    bfr[kc].q[1] = *reinterpret_cast<const uint4*>(p + 16 + 8 * kg);
  }

  CFrag acc = {};
  for (int h = 0; h < 16; ++h) {
    __syncthreads();  // all waves done with qis of previous head
#if HAVE_TDM
    if (tid == 0) {
      tdm_load_2d_bf16(qi + ((size_t)b * S_ + s0) * 2048 + h * 128,
                       lds_byte_addr(qis),
                       /*dim0=*/128u, /*dim1=*/16u,
                       /*tile0=*/128u, /*tile1=*/16u, /*stride0=*/2048ull);
      __builtin_amdgcn_s_wait_tensorcnt(0);
    }
#else
    {  // stage qi for this head
      const int sr = tid >> 4;
      const int dc = (tid & 15) * 8;
      const bf16* src = qi + ((size_t)b * S_ + s0 + sr) * 2048 + h * 128 + dc;
      *reinterpret_cast<uint4*>(qis + sr * 128 + dc) =
          *reinterpret_cast<const uint4*>(src);
    }
#endif
    __syncthreads();

    CFrag sc = {};
#pragma unroll
    for (int kc = 0; kc < 4; ++kc) {
      AFrag afr;
      const bf16* p = qis + l15 * 128 + kc * 32;
      afr.q[0] = *reinterpret_cast<const uint4*>(p + 8 * kg);
      afr.q[1] = *reinterpret_cast<const uint4*>(p + 16 + 8 * kg);
      sc.v = wmma_bf16(afr.v, bfr[kc].v, sc.v);
    }
#pragma unroll
    for (int r = 0; r < 8; ++r) {
      const float v = fmaxf(sc.e[r] * 0.08838834764831845f, 0.0f);  // IDX_D^-0.5
      acc.e[r] += v * wl[(r + 8 * kg) * 16 + h];
    }
  }

#pragma unroll
  for (int r = 0; r < 8; ++r) {
    const int sg = s0 + r + 8 * kg;
    const int tg = t0 + w * 16 + l15;
    const float v = acc.e[r] + ((tg <= sg) ? 0.0f : kNEG);
    out[((size_t)b * S_ + sg) * S_ + tg] = v;
  }
}

// ---------------------------------------------------------------------------
// Top-k (512 of 2048) per row via 4x8-bit radix select; ties taken in index
// order (matches jax top_k). Output: bitmask u32[64] per row (1 = kept).
// ---------------------------------------------------------------------------
__global__ __launch_bounds__(256)
void topk_kernel(const float* __restrict__ scores, unsigned* __restrict__ maskout)
{
  const int row = blockIdx.x;  // b*S + s
  const int tid = threadIdx.x;
  const float* src = scores + (size_t)row * S_;

  __shared__ unsigned hist[256];
  __shared__ unsigned char flags[2048];
  __shared__ unsigned selw[64];
  __shared__ unsigned sh_prefix, sh_krem;

  unsigned key[8];
#pragma unroll
  for (int i = 0; i < 8; ++i) {
    const unsigned u = __float_as_uint(src[i * 256 + tid]);
    key[i] = (u & 0x80000000u) ? ~u : (u | 0x80000000u);  // monotone key
  }

  unsigned prefix = 0;
  for (int pass = 0; pass < 4; ++pass) {
    const int shift = 24 - pass * 8;
    hist[tid] = 0; __syncthreads();
#pragma unroll
    for (int i = 0; i < 8; ++i) {
      const bool act = (pass == 0) || ((key[i] >> (shift + 8)) == prefix);
      if (act) atomicAdd(&hist[(key[i] >> shift) & 255u], 1u);
    }
    __syncthreads();
    if (tid == 0) {
      const unsigned krem = (pass == 0) ? (unsigned)TOPK_ : sh_krem;
      unsigned cum = 0;
      unsigned bsel = 0, newkrem = krem;
      for (int bv = 255; bv >= 0; --bv) {
        cum += hist[bv];
        if (cum >= krem) { bsel = (unsigned)bv; newkrem = krem - (cum - hist[bv]); break; }
      }
      sh_prefix = (prefix << 8) | bsel;
      sh_krem = newkrem;
    }
    __syncthreads();
    prefix = sh_prefix;
  }

  // prefix == threshold key; sh_krem == #elements equal to threshold to keep
#pragma unroll
  for (int i = 0; i < 8; ++i) {
    const unsigned char f = (key[i] > prefix) ? 2 : ((key[i] == prefix) ? 1 : 0);
    flags[i * 256 + tid] = f;
  }
  if (tid < 64) selw[tid] = 0;
  __syncthreads();
  if (tid == 0) {
    unsigned taken = 0;
    const unsigned krem = sh_krem;
    for (int t = 0; t < 2048; ++t) {
      const unsigned char f = flags[t];
      bool sel = (f == 2);
      if (f == 1 && taken < krem) { sel = true; ++taken; }
      if (sel) selw[t >> 5] |= (1u << (t & 31));
    }
  }
  __syncthreads();
  if (tid < 64) maskout[(size_t)row * 64 + tid] = selw[tid];
}

// ---------------------------------------------------------------------------
// Flash attention with combined causal + top-k mask.
// Block: 128 s-rows (8 waves x 16 rows), fixed (b,h); loop t in 32-tiles.
// K tile staged via TDM when available; V tile transposed manually.
// ---------------------------------------------------------------------------
__global__ __launch_bounds__(256)
void attention_kernel(const bf16* __restrict__ query, const bf16* __restrict__ keyt,
                      const bf16* __restrict__ value, const unsigned* __restrict__ selmask,
                      float* __restrict__ attn_out)
{
  constexpr float SC = 0.07216878364870323f;  // QK_HEAD^-0.5
  const int b = blockIdx.z, h = blockIdx.y;
  const int s_base = blockIdx.x * 128;
  const int tid = threadIdx.x;
  const int lane = tid & 31;
  const int w = tid >> 5;
  const int l15 = lane & 15;
  const int kg = lane >> 4;

  __shared__ bf16 Ks[32 * 192];   // [t][d]
  __shared__ bf16 Vst[128 * 32];  // transposed: [d][t]
  __shared__ bf16 Ps[8][16 * 32]; // per-wave P scratch [m][t]

  const int sw = s_base + w * 16;
  const size_t khbase = ((size_t)b * H_ + h) * S_;

  // preload Q fragments (held for whole kernel)
  AFrag qf[6];
  {
    const bf16* qrow = query + (khbase + sw + l15) * 192;
#pragma unroll
    for (int kc = 0; kc < 6; ++kc) {
      qf[kc].q[0] = *reinterpret_cast<const uint4*>(qrow + kc * 32 + 8 * kg);
      qf[kc].q[1] = *reinterpret_cast<const uint4*>(qrow + kc * 32 + 16 + 8 * kg);
    }
  }

  CFrag out[8] = {};
  float rm[8], rs[8];
#pragma unroll
  for (int r = 0; r < 8; ++r) { rm[r] = -1e30f; rs[r] = 0.0f; }

  const int tend = s_base + 128;
  for (int t0 = 0; t0 < tend; t0 += 32) {
#if HAVE_TDM
    if (tid == 0) {
      tdm_load_2d_bf16(keyt + (khbase + t0) * 192, lds_byte_addr(Ks),
                       /*dim0=*/192u, /*dim1=*/(unsigned)(S_ - t0),
                       /*tile0=*/192u, /*tile1=*/32u, /*stride0=*/192ull);
      __builtin_amdgcn_s_wait_tensorcnt(0);
    }
#else
    {  // stage K tile 32x192
      const int tr = tid >> 3;
      const int dc = (tid & 7) * 24;
      const bf16* src = keyt + (khbase + t0 + tr) * 192 + dc;
      bf16* dst = Ks + tr * 192 + dc;
#pragma unroll
      for (int i = 0; i < 24; i += 8)
        *reinterpret_cast<uint4*>(dst + i) = *reinterpret_cast<const uint4*>(src + i);
    }
#endif
    {  // stage V tile transposed -> Vst[d][t]
      const int tr = tid >> 3;
      const int dc = (tid & 7) * 16;
      const bf16* src = value + (khbase + t0 + tr) * 128 + dc;
#pragma unroll
      for (int i = 0; i < 16; ++i) Vst[(dc + i) * 32 + tr] = src[i];
    }
    __syncthreads();

    // scores: 16x32 per wave (two 16x16 halves), K=192
    CFrag scf[2];
#pragma unroll
    for (int th = 0; th < 2; ++th) {
#pragma unroll
      for (int r = 0; r < 8; ++r) scf[th].e[r] = 0.0f;
#pragma unroll
      for (int kc = 0; kc < 6; ++kc) {
        AFrag bf_;
        const bf16* p = Ks + (th * 16 + l15) * 192 + kc * 32;
        bf_.q[0] = *reinterpret_cast<const uint4*>(p + 8 * kg);
        bf_.q[1] = *reinterpret_cast<const uint4*>(p + 16 + 8 * kg);
        scf[th].v = wmma_bf16(qf[kc].v, bf_.v, scf[th].v);
      }
    }

    // scale + causal + top-k mask; track tile max
    float tmax[8];
#pragma unroll
    for (int r = 0; r < 8; ++r) tmax[r] = -3e38f;
#pragma unroll
    for (int th = 0; th < 2; ++th) {
#pragma unroll
      for (int r = 0; r < 8; ++r) {
        const int tg = t0 + th * 16 + l15;
        const int sg = sw + r + 8 * kg;
        float v = scf[th].e[r] * SC;
        if (tg > sg) v += kNEG;
        const unsigned wbit = selmask[((size_t)b * S_ + sg) * 64 + (tg >> 5)];
        if (!((wbit >> (tg & 31)) & 1u)) v += kNEG;
        scf[th].e[r] = v;
        tmax[r] = fmaxf(tmax[r], v);
      }
    }
#pragma unroll
    for (int r = 0; r < 8; ++r) {
      tmax[r] = fmaxf(tmax[r], __shfl_xor(tmax[r], 1, 32));
      tmax[r] = fmaxf(tmax[r], __shfl_xor(tmax[r], 2, 32));
      tmax[r] = fmaxf(tmax[r], __shfl_xor(tmax[r], 4, 32));
      tmax[r] = fmaxf(tmax[r], __shfl_xor(tmax[r], 8, 32));
    }
    // online softmax update + write P to LDS (bf16)
#pragma unroll
    for (int r = 0; r < 8; ++r) {
      const float nm = fmaxf(rm[r], tmax[r]);
      const float so = __expf(rm[r] - nm);
      rm[r] = nm;
      float ps = 0.0f;
#pragma unroll
      for (int th = 0; th < 2; ++th) {
        const float p = __expf(scf[th].e[r] - nm);
        ps += p;
        Ps[w][(r + 8 * kg) * 32 + th * 16 + l15] = (bf16)p;
      }
      ps += __shfl_xor(ps, 1, 32);
      ps += __shfl_xor(ps, 2, 32);
      ps += __shfl_xor(ps, 4, 32);
      ps += __shfl_xor(ps, 8, 32);
      rs[r] = rs[r] * so + ps;
#pragma unroll
      for (int ds = 0; ds < 8; ++ds) out[ds].e[r] *= so;
    }
    __syncthreads();

    // P(16x32) x V(32x128)
    AFrag pf;
    {
      const bf16* p = Ps[w] + l15 * 32;
      pf.q[0] = *reinterpret_cast<const uint4*>(p + 8 * kg);
      pf.q[1] = *reinterpret_cast<const uint4*>(p + 16 + 8 * kg);
    }
#pragma unroll
    for (int ds = 0; ds < 8; ++ds) {
      AFrag vf;
      const bf16* p = Vst + (ds * 16 + l15) * 32;
      vf.q[0] = *reinterpret_cast<const uint4*>(p + 8 * kg);
      vf.q[1] = *reinterpret_cast<const uint4*>(p + 16 + 8 * kg);
      out[ds].v = wmma_bf16(pf.v, vf.v, out[ds].v);
    }
    __syncthreads();
  }

#pragma unroll
  for (int ds = 0; ds < 8; ++ds)
#pragma unroll
    for (int r = 0; r < 8; ++r) {
      const int sg = sw + r + 8 * kg;
      const int d = ds * 16 + l15;
      attn_out[((size_t)b * S_ + sg) * 2048 + h * 128 + d] = out[ds].e[r] / rs[r];
    }
}

// ---------------------------------------------------------------------------
// Host-side orchestration
// ---------------------------------------------------------------------------
extern "C" void kernel_launch(void* const* d_in, const int* in_sizes, int n_in,
                              void* d_out, int out_size, void* d_ws, size_t ws_size,
                              hipStream_t stream)
{
  const float* hs       = (const float*)d_in[0];   // [B,S,HID]
  const float* cosb     = (const float*)d_in[1];   // [B,S,64]
  const float* sinb     = (const float*)d_in[2];   // [B,S,64]
  // d_in[3] = attention_mask (causal; recomputed on device)
  const float* Wqa      = (const float*)d_in[4];   // [HID, Q_RANK]
  const float* rms_qa_w = (const float*)d_in[5];
  const float* Wqb      = (const float*)d_in[6];   // [Q_RANK, H*192]
  const float* Wkva     = (const float*)d_in[7];   // [HID, 576]
  const float* rms_kva_w= (const float*)d_in[8];
  const float* Wkvb     = (const float*)d_in[9];   // [512, H*256]
  const float* Wo       = (const float*)d_in[10];  // [H*128, HID]
  const float* idx_wqb  = (const float*)d_in[11];  // [Q_RANK, 16*128]
  const float* idx_wk   = (const float*)d_in[12];  // [HID, 128]
  const float* idx_ln_w = (const float*)d_in[13];
  const float* idx_ln_b = (const float*)d_in[14];
  const float* idx_wproj= (const float*)d_in[15];  // [HID, 16]

  const size_t BS = (size_t)B_ * S_;  // 4096
  char* ws = (char*)d_ws;
  size_t off = 0;
  auto take = [&](size_t bytes) -> char* {
    char* p = ws + off;
    off += (bytes + 255) & ~(size_t)255;
    return p;
  };

  float* q_resid = (float*)take(BS * Q_RANK_ * 4);
  float* q_full  = (float*)take(BS * H_ * QK_HEAD_ * 4);  // aliased by idx_scores
  float* ckv     = (float*)take(BS * 576 * 4);
  float* k_comp  = (float*)take(BS * 512 * 4);
  float* kpe     = (float*)take(BS * 64 * 4);
  float* kv_full = (float*)take(BS * 4096 * 4);           // aliased by attn_out
  bf16*  query   = (bf16*)take(BS * H_ * 192 * 2);
  bf16*  keyt    = (bf16*)take(BS * H_ * 192 * 2);
  bf16*  value   = (bf16*)take(BS * H_ * 128 * 2);
  float* qi_full = (float*)take(BS * 2048 * 4);
  bf16*  qi      = (bf16*)take(BS * 2048 * 2);
  float* ki_tmp  = (float*)take(BS * 128 * 4);
  bf16*  ki      = (bf16*)take(BS * 128 * 2);
  float* wts     = (float*)take(BS * 16 * 4);
  unsigned* selm = (unsigned*)take(BS * 64 * 4);
  float* idx_scores = q_full;   // reuse after query built
  float* attn_out   = kv_full;  // reuse after keyt/value built

  const int M = (int)BS;
  auto gemm = [&](const float* A, const float* Bm, float* C, int N, int K) {
    dim3 grid((N + 127) / 128, M / 128);
    gemm_f32_bf16wmma<<<grid, 256, 0, stream>>>(A, Bm, C, M, N, K);
  };

  // 1) q_resid = rmsnorm(hs @ Wqa)
  gemm(hs, Wqa, q_resid, Q_RANK_, HID_);
  rmsnorm_kernel<<<M, 256, 0, stream>>>(q_resid, rms_qa_w, q_resid,
                                        Q_RANK_, Q_RANK_, Q_RANK_);
  // 2) q_full = q_resid @ Wqb
  gemm(q_resid, Wqb, q_full, H_ * QK_HEAD_, Q_RANK_);
  // 3) ckv = hs @ Wkva; split: rmsnorm + rope(k_pe)
  gemm(hs, Wkva, ckv, KV_RANK_ + ROPE_, HID_);
  rmsnorm_kernel<<<M, 256, 0, stream>>>(ckv, rms_kva_w, k_comp, KV_RANK_, 576, 512);
  rope_kpe_kernel<<<M, 64, 0, stream>>>(ckv, cosb, sinb, kpe);
  // 4) kv_full = k_comp @ Wkvb
  gemm(k_comp, Wkvb, kv_full, H_ * (QK_NOPE_ + V_HEAD_), KV_RANK_);
  // 5) build bf16 Q/K/V tensors (RoPE applied)
  build_query_kernel<<<M * H_, 192, 0, stream>>>(q_full, cosb, sinb, query);
  build_keyval_kernel<<<M * H_, 256, 0, stream>>>(kv_full, kpe, keyt, value);
  // 6) indexer projections
  gemm(q_resid, idx_wqb, qi_full, IDX_H_ * IDX_D_, Q_RANK_);
  gemm(hs, idx_wk, ki_tmp, IDX_D_, HID_);
  gemm(hs, idx_wproj, wts, IDX_H_, HID_);
  build_qi_kernel<<<M * IDX_H_, 128, 0, stream>>>(qi_full, cosb, sinb, qi);
  build_ki_kernel<<<M, 128, 0, stream>>>(ki_tmp, idx_ln_w, idx_ln_b, cosb, sinb, ki);
  // 7) index scores (fused relu-weighted head reduction + causal)
  {
    dim3 grid(S_ / 128, S_ / 16, B_);
    idx_scores_kernel<<<grid, 256, 0, stream>>>(qi, ki, wts, idx_scores);
  }
  // 8) top-k -> bitmask
  topk_kernel<<<M, 256, 0, stream>>>(idx_scores, selm);
  // 9) flash attention with combined mask
  {
    dim3 grid(S_ / 128, H_, B_);
    attention_kernel<<<grid, 256, 0, stream>>>(query, keyt, value, selm, attn_out);
  }
  // 10) output projection
  gemm(attn_out, Wo, (float*)d_out, HID_, H_ * V_HEAD_);
  (void)in_sizes; (void)n_in; (void)out_size; (void)ws_size;
}
